// ASRGCN_66322884985191
// MI455X (gfx1250) — matrix operand order repaired and checked by reading, not compiled, not run on hardware
//
#include <hip/hip_runtime.h>
#include <hip/hip_bf16.h>

// ASRGCN forward for MI455X (gfx1250, wave32, WMMA).
//  - bf16 inputs, f32 accumulate via v_wmma_f32_16x16x32_bf16.
//  - LSTM input projection hoisted to one big WMMA GEMM (Xpre).
//  - LSTM recurrence: 8 persistent blocks (4 batch tiles x 2 dirs), h/c/g in LDS.
//  - adj is 512-nnz/batch sparse -> relation-list aggregation (no dense adj matmul),
//    deterministic (int atomics only, fixed-order float sums).
//  - bilinear score via precomputed relM = rel_embed @ bil_W^T.
//  - K-loop fragments are double-buffered so global_load_b128 for step k+1 issue
//    before the WMMAs of step k (avoids s_wait_loadcnt 0 stalls per WMMA).

typedef __bf16 bf16_t;
typedef __attribute__((ext_vector_type(8)))  __bf16 v8bf;
typedef __attribute__((ext_vector_type(16))) __bf16 v16bf;
typedef __attribute__((ext_vector_type(8)))  float  v8f;

#define BB   64
#define SS   256
#define EE   300
#define HH   300
#define RR   512
#define NRELD 1200   // 4H
#define KE   320     // padded E / H (K of lstm GEMMs)
#define N4H  1200
#define N4HP 1216
#define N2H  600
#define K2H  640     // padded 2H as K
#define N2HP 640

__device__ __forceinline__ float sigmf(float x) { return 1.f / (1.f + __expf(-x)); }

// A fragment (16x32 bf16, MxK): lane = M row (lane&15); kb0 = (lane>=16)?16:0.
// elements 0..15 <- p[0..15] contiguous (p already includes kb0).
__device__ __forceinline__ v16bf load_a_frag(const bf16_t* p) {
  v8bf lo = *(const v8bf*)p;
  v8bf hi = *(const v8bf*)(p + 8);
  v16bf r;
#pragma unroll
  for (int i = 0; i < 8; ++i) { r[i] = lo[i]; r[i + 8] = hi[i]; }
  return r;
}
// B fragment (32x16 bf16, KxN), N-major storage: lane = N col (lane&15);
// kb = (lane>=16)?16:0; elements 0..15 <- p[0..15] contiguous (p includes kb).
__device__ __forceinline__ v16bf load_b_frag(const bf16_t* p) {
  v8bf lo = *(const v8bf*)p;
  v8bf hi = *(const v8bf*)(p + 8);
  v16bf r;
#pragma unroll
  for (int i = 0; i < 8; ++i) { r[i] = lo[i]; r[i + 8] = hi[i]; }
  return r;
}

// ---------------- prep kernels ----------------

// Convert f32 weight to bf16, zero-padded, optional transpose.
// transpose==0: out[n*Kp+k] = W[n*Ks+k]      (W is [Ns,Ks], already N-major)
// transpose==1: out[n*Kp+k] = W[k*Ns+n]      (W is [Ks,Ns] -> N-major of W^T)
__global__ __launch_bounds__(256)
void prep_w_kernel(const float* __restrict__ W, bf16_t* __restrict__ out,
                   int Ns, int Ks, int Np, int Kp, int transpose) {
  long i = (long)blockIdx.x * 256 + threadIdx.x;
  long tot = (long)Np * Kp;
  if (i >= tot) return;
  int n = (int)(i / Kp), k = (int)(i - (long)n * Kp);
  float v = 0.f;
  if (n < Ns && k < Ks) v = transpose ? W[(long)k * Ns + n] : W[(long)n * Ks + k];
  out[i] = (bf16_t)v;
}

// relM[rel][c] = sum_j bil_W[c][j] * rel_embed[rel][j]   (46 x 1200)
__global__ __launch_bounds__(256)
void relm_kernel(const float* __restrict__ rel_embed, const float* __restrict__ bil_W,
                 float* __restrict__ relM) {
  int i = blockIdx.x * 256 + threadIdx.x;
  if (i >= 46 * NRELD) return;
  int rel = i / NRELD, c = i - rel * NRELD;
  float s = 0.f;
  for (int j = 0; j < 50; ++j) s += bil_W[c * 50 + j] * rel_embed[rel * 50 + j];
  relM[i] = s;
}

// per-batch lengths; lensf[b*4+{0,1,2,3}] = {l0, l1, ctx, text_len}; lensi = {l0,l1}
__global__ void lens_kernel(const int* __restrict__ text, const int* __restrict__ aspect,
                            const int* __restrict__ left, float* __restrict__ lensf,
                            int* __restrict__ lensi) {
  int b = threadIdx.x;
  if (b >= BB) return;
  int tl = 0, al = 0, ll = 0;
  for (int s = 0; s < SS; ++s) tl += (text[b * SS + s] != 0);
  for (int i = 0; i < 4; ++i)  al += (aspect[b * 4 + i] != 0);
  for (int i = 0; i < 64; ++i) ll += (left[b * 64 + i] != 0);
  lensf[b * 4 + 0] = (float)ll;
  lensf[b * 4 + 1] = (float)(ll + al - 1);
  lensf[b * 4 + 2] = (float)(tl - al);
  lensf[b * 4 + 3] = (float)tl;
  lensi[b * 2 + 0] = ll;
  lensi[b * 2 + 1] = ll + al - 1;
}

// embedding gather -> bf16 padded [16384, KE]
__global__ __launch_bounds__(256)
void embed_kernel(const int* __restrict__ idx, const float* __restrict__ table,
                  bf16_t* __restrict__ out) {
  long i = (long)blockIdx.x * 256 + threadIdx.x;
  if (i >= (long)BB * SS * KE) return;
  int e = (int)(i % KE);
  long row = i / KE;
  float v = 0.f;
  if (e < EE) v = table[(long)idx[row] * EE + e];
  out[i] = (bf16_t)v;
}

// ---------------- generic WMMA GEMM: C[M,N] = A[M,Kp] * Bn[N-major,Kp]^T (+bias) ----
// grid.x = M/16, grid.y = ceil(((N+63)&~63)/512), block = 256 (8 waves),
// each wave computes a 16x64 output macro-tile (A fragment reused 4x).
// Fragments double-buffered: loads for K-step k+1 issue before WMMAs of step k.
__global__ __launch_bounds__(256)
void gemm_bf16_wmma(const bf16_t* __restrict__ A, int Kp,
                    const bf16_t* __restrict__ Bn,
                    float* __restrict__ C, int N,
                    const float* __restrict__ bias) {
  const int lane = threadIdx.x & 31;
  const int wave = threadIdx.x >> 5;
  const int npad64 = (N + 63) & ~63;
  const int nbase = (blockIdx.y * 8 + wave) * 64;
  if (nbase >= npad64) return;           // wave-uniform
  const int ln = lane & 15;
  const int lh = lane >> 4;
  const bf16_t* ap  = A  + (size_t)(blockIdx.x * 16 + ln) * Kp + (lh << 4);
  const bf16_t* bp0 = Bn + (size_t)(nbase + ln) * Kp + (lh << 4);
  v8f acc[4];
#pragma unroll
  for (int t = 0; t < 4; ++t)
#pragma unroll
    for (int r = 0; r < 8; ++r) acc[t][r] = 0.f;

  v16bf a0 = load_a_frag(ap);
  v16bf b0[4];
#pragma unroll
  for (int t = 0; t < 4; ++t) b0[t] = load_b_frag(bp0 + (size_t)t * 16 * Kp);

  for (int kt = 0; kt < Kp; kt += 32) {
    const int kn = (kt + 32 < Kp) ? kt + 32 : kt;  // last iter: harmless reload
    v16bf a1 = load_a_frag(ap + kn);
    v16bf b1[4];
#pragma unroll
    for (int t = 0; t < 4; ++t) b1[t] = load_b_frag(bp0 + (size_t)t * 16 * Kp + kn);
#pragma unroll
    for (int t = 0; t < 4; ++t)
      acc[t] = __builtin_amdgcn_wmma_f32_16x16x32_bf16(
          false, a0, false, b0[t], (short)0, acc[t], false, false);
    a0 = a1;
#pragma unroll
    for (int t = 0; t < 4; ++t) b0[t] = b1[t];
  }

  const int rbase = blockIdx.x * 16 + (lh << 3);
#pragma unroll
  for (int t = 0; t < 4; ++t) {
    int gc = nbase + t * 16 + ln;
    if (gc < N) {
      float bv = bias ? bias[gc] : 0.f;
#pragma unroll
      for (int r = 0; r < 8; ++r)
        C[(size_t)(rbase + r) * N + gc] = acc[t][r] + bv;
    }
  }
}

// ---------------- LSTM recurrence ----------------
// grid = (4 batch tiles, 2 dirs), block = 512 (16 waves).
// g = Xpre[t] + h @ Whh^T computed with WMMA (h bf16 in LDS), then gate math.
// Global Whh fragments are prefetched one K-step ahead; A comes from LDS.
__global__ __launch_bounds__(512)
void lstm_rec_kernel(const float* __restrict__ Xf, const float* __restrict__ Xb,
                     const bf16_t* __restrict__ Wf, const bf16_t* __restrict__ Wb,
                     float* __restrict__ text_out) {
  extern __shared__ char smem[];
  float*  g_s  = (float*)smem;                                  // 16*1200 f32
  float*  c_s  = (float*)(smem + 16 * N4H * 4);                 // 16*300  f32
  bf16_t* h_bf = (bf16_t*)(smem + 16 * N4H * 4 + 16 * HH * 4);  // 16*320  bf16
  const int dir = blockIdx.y;
  const int bm  = blockIdx.x * 16;
  const float*  Xpre = dir ? Xb : Xf;
  const bf16_t* Whh  = dir ? Wb : Wf;
  const int tid = threadIdx.x;
  const int lane = tid & 31, wave = tid >> 5;
  const int ln = lane & 15, lh = lane >> 4;

  for (int i = tid; i < 16 * KE; i += 512) h_bf[i] = (bf16_t)0.f;
  for (int i = tid; i < 16 * HH; i += 512) c_s[i] = 0.f;
  __syncthreads();

  for (int step = 0; step < SS; ++step) {
    const int s = dir ? (SS - 1 - step) : step;
    for (int nt = wave; nt < 75; nt += 16) {   // wave-uniform trip count
      const int col = nt * 16 + ln;
      v8f acc;
#pragma unroll
      for (int r = 0; r < 8; ++r)
        acc[r] = Xpre[((size_t)(bm + r + (lh << 3)) * SS + s) * N4H + col];
      const bf16_t* hp = h_bf + ln * KE + (lh << 4);
      const bf16_t* bp = Whh + (size_t)col * KE + (lh << 4);
      v16bf b0 = load_b_frag(bp);
      for (int kt = 0; kt < KE; kt += 32) {
        const int kn = (kt + 32 < KE) ? kt + 32 : kt;
        v16bf b1 = load_b_frag(bp + kn);        // prefetch next global frag
        v16bf a = load_a_frag(hp + kt);         // LDS
        acc = __builtin_amdgcn_wmma_f32_16x16x32_bf16(
            false, a, false, b0, (short)0, acc, false, false);
        b0 = b1;
      }
#pragma unroll
      for (int r = 0; r < 8; ++r)
        g_s[(r + (lh << 3)) * N4H + col] = acc[r];
    }
    __syncthreads();
    for (int idx = tid; idx < 16 * HH; idx += 512) {
      const int m = idx / HH, j = idx - m * HH;
      const float gi = g_s[m * N4H + j];
      const float gf = g_s[m * N4H + HH + j];
      const float gg = g_s[m * N4H + 2 * HH + j];
      const float go = g_s[m * N4H + 3 * HH + j];
      float c = sigmf(gf) * c_s[idx] + sigmf(gi) * tanhf(gg);
      float h = sigmf(go) * tanhf(c);
      c_s[idx] = c;
      h_bf[m * KE + j] = (bf16_t)h;
      text_out[((size_t)(bm + m) * SS + s) * N2H + dir * HH + j] = h;
    }
    __syncthreads();
  }
}

// ---------------- relation scores ----------------
// score[b,r] = sigmoid(text_out[b,head].relM[rel][:600] +
//                      text_out[b,behead].relM[rel][600:] + bil_b)
__global__ __launch_bounds__(256)
void relscore_kernel(const float* __restrict__ text_out, const float* __restrict__ relM,
                     const int* __restrict__ head, const int* __restrict__ behead,
                     const int* __restrict__ relv, const float* __restrict__ bil_b,
                     float* __restrict__ score) {
  int gw = (int)((blockIdx.x * 256 + threadIdx.x) >> 5);
  if (gw >= BB * RR) return;
  int lane = threadIdx.x & 31;
  int b = gw >> 9, r = gw & (RR - 1);
  int rel = relv[b * RR + r];
  float s = 0.f;
  if (rel != 0) {
    const float* n1 = text_out + ((size_t)b * SS + head[b * RR + r]) * N2H;
    const float* n2 = text_out + ((size_t)b * SS + behead[b * RR + r]) * N2H;
    const float* m  = relM + (size_t)rel * NRELD;
    for (int c = lane; c < N2H; c += 32) s += n1[c] * m[c] + n2[c] * m[N2H + c];
  }
#pragma unroll
  for (int off = 16; off; off >>= 1) s += __shfl_xor(s, off, 32);
  if (lane == 0) score[b * RR + r] = sigmf(s + bil_b[0]);
}

// row ranges of the sorted relation list (deterministic int atomics)
__global__ void rowinit_kernel(int* __restrict__ first, int* __restrict__ cnt) {
  int i = blockIdx.x * 256 + threadIdx.x;
  if (i >= BB * SS) return;
  first[i] = 1 << 30;
  cnt[i] = 0;
}
__global__ void rowbuild_kernel(const int* __restrict__ head, int* __restrict__ first,
                                int* __restrict__ cnt) {
  int i = blockIdx.x * 256 + threadIdx.x;
  if (i >= BB * RR) return;
  int b = i >> 9;
  int row = b * SS + head[i];
  atomicMin(&first[row], i & (RR - 1));
  atomicAdd(&cnt[row], 1);
}

// position weight + bf16 pad -> [16384, K2H]
__global__ __launch_bounds__(256)
void poswt_kernel(const float* __restrict__ X, const float* __restrict__ lensf,
                  bf16_t* __restrict__ out) {
  long i = (long)blockIdx.x * 256 + threadIdx.x;
  if (i >= (long)BB * SS * K2H) return;
  int e = (int)(i % K2H);
  long row = i / K2H;
  float v = 0.f;
  if (e < N2H) {
    int b = (int)(row >> 8), s = (int)(row & (SS - 1));
    float l0 = lensf[b * 4], l1 = lensf[b * 4 + 1], ctx = lensf[b * 4 + 2],
          tl = lensf[b * 4 + 3];
    float sj = (float)s;
    float w = (sj < l0) ? (1.f - (l0 - sj) / ctx)
            : (sj <= l1) ? 0.f
            : (sj < tl)  ? (1.f - (sj - l1) / ctx) : 0.f;
    v = w * X[row * N2H + e];
  }
  out[i] = (bf16_t)v;
}

// GCN aggregation + finalize: out = relu((adj@hidden)/denom + bias)
__global__ __launch_bounds__(256)
void gcn_agg_kernel(const float* __restrict__ hidden, const float* __restrict__ score,
                    const int* __restrict__ behead, const int* __restrict__ first,
                    const int* __restrict__ cnt, const float* __restrict__ bias,
                    float* __restrict__ out) {
  long i = (long)blockIdx.x * 256 + threadIdx.x;
  if (i >= (long)BB * SS * N2H) return;
  int col = (int)(i % N2H);
  long row = i / N2H;
  int b = (int)(row >> 8);
  int f = first[row], n = cnt[row];
  float acc = 0.f, den = 1.f;
  for (int k = 0; k < n; ++k) {
    int r = f + k;
    float sc = score[b * RR + r];
    int bh = behead[b * RR + r];
    acc += sc * hidden[((size_t)b * SS + bh) * N2H + col];
    den += sc;
  }
  float v = acc / den + bias[col];
  out[i] = v > 0.f ? v : 0.f;
}

// ---------------- attention + FC ----------------
__global__ void attn_sum_kernel(const float* __restrict__ x2, const int* __restrict__ lensi,
                                float* __restrict__ sumx) {
  int i = blockIdx.x * 256 + threadIdx.x;
  if (i >= BB * N2H) return;
  int b = i / N2H, col = i - b * N2H;
  int l0 = lensi[b * 2], l1 = lensi[b * 2 + 1];
  if (l0 < 0) l0 = 0;
  if (l1 > SS - 1) l1 = SS - 1;
  float s = 0.f;
  for (int t = l0; t <= l1; ++t) s += x2[((size_t)b * SS + t) * N2H + col];
  sumx[i] = s;
}
__global__ void attn_score_kernel(const float* __restrict__ sumx,
                                  const float* __restrict__ text_out,
                                  float* __restrict__ tsc) {
  int i = blockIdx.x * 256 + threadIdx.x;
  if (i >= BB * SS) return;
  int b = i >> 8;
  float s = 0.f;
  for (int c = 0; c < N2H; ++c) s += sumx[b * N2H + c] * text_out[(size_t)i * N2H + c];
  tsc[i] = s;
}
__global__ __launch_bounds__(256)
void softmax_kernel(const float* __restrict__ tsc, float* __restrict__ alpha) {
  __shared__ float red[256];
  int b = blockIdx.x, t = threadIdx.x;
  float v = tsc[b * SS + t];
  red[t] = v;
  __syncthreads();
  for (int off = 128; off; off >>= 1) {
    if (t < off) red[t] = fmaxf(red[t], red[t + off]);
    __syncthreads();
  }
  float mx = red[0];
  __syncthreads();
  float e = __expf(v - mx);
  red[t] = e;
  __syncthreads();
  for (int off = 128; off; off >>= 1) {
    if (t < off) red[t] += red[t + off];
    __syncthreads();
  }
  alpha[b * SS + t] = e / red[0];
}
__global__ void attn_out_kernel(const float* __restrict__ alpha,
                                const float* __restrict__ text_out,
                                float* __restrict__ out2h) {
  int i = blockIdx.x * 256 + threadIdx.x;
  if (i >= BB * N2H) return;
  int b = i / N2H, c = i - b * N2H;
  float s = 0.f;
  for (int t = 0; t < SS; ++t) s += alpha[b * SS + t] * text_out[((size_t)b * SS + t) * N2H + c];
  out2h[i] = s;
}
__global__ void fc_kernel(const float* __restrict__ out2h, const float* __restrict__ fc_W,
                          const float* __restrict__ fc_b, float* __restrict__ out) {
  int i = blockIdx.x * 256 + threadIdx.x;
  if (i >= BB * 3) return;
  int b = i / 3, p = i - b * 3;
  float s = fc_b[p];
  for (int c = 0; c < N2H; ++c) s += out2h[b * N2H + c] * fc_W[c * 3 + p];
  out[i] = s;
}

// ---------------- host launch ----------------
extern "C" void kernel_launch(void* const* d_in, const int* in_sizes, int n_in,
                              void* d_out, int out_size, void* d_ws, size_t ws_size,
                              hipStream_t stream) {
  (void)in_sizes; (void)n_in; (void)out_size; (void)ws_size;
  const int*   text_idx   = (const int*)d_in[0];
  const int*   aspect_idx = (const int*)d_in[1];
  const int*   left_idx   = (const int*)d_in[2];
  // d_in[3] = dense adj: unused (reconstructed sparsely from head/behead/score)
  const int*   head_v     = (const int*)d_in[4];
  const int*   behead_v   = (const int*)d_in[5];
  const int*   rel_v      = (const int*)d_in[6];
  const float* embed_tab  = (const float*)d_in[7];
  const float* rel_embed  = (const float*)d_in[8];
  const float* Wf_ih = (const float*)d_in[9];
  const float* Wf_hh = (const float*)d_in[10];
  const float* bf    = (const float*)d_in[11];
  const float* Wb_ih = (const float*)d_in[12];
  const float* Wb_hh = (const float*)d_in[13];
  const float* bb    = (const float*)d_in[14];
  const float* bil_W = (const float*)d_in[15];
  const float* bil_b = (const float*)d_in[16];
  const float* gc1_W = (const float*)d_in[17];
  const float* gc1_b = (const float*)d_in[18];
  const float* gc2_W = (const float*)d_in[19];
  const float* gc2_b = (const float*)d_in[20];
  const float* fc_W  = (const float*)d_in[21];
  const float* fc_b  = (const float*)d_in[22];
  float* out = (float*)d_out;

  char* w = (char*)d_ws;
  size_t off = 0;
  auto take = [&](size_t bytes) -> char* {
    char* p = w + off;
    off += (bytes + 255) & ~(size_t)255;
    return p;
  };
  bf16_t* textbf = (bf16_t*)take((size_t)BB * SS * KE * 2);
  bf16_t* wfih = (bf16_t*)take((size_t)N4HP * KE * 2);
  bf16_t* wfhh = (bf16_t*)take((size_t)N4HP * KE * 2);
  bf16_t* wbih = (bf16_t*)take((size_t)N4HP * KE * 2);
  bf16_t* wbhh = (bf16_t*)take((size_t)N4HP * KE * 2);
  bf16_t* gc1w = (bf16_t*)take((size_t)N2HP * K2H * 2);
  bf16_t* gc2w = (bf16_t*)take((size_t)N2HP * K2H * 2);
  float*  relM = (float*)take((size_t)46 * NRELD * 4);
  float*  lensf = (float*)take(BB * 4 * 4);
  int*    lensi = (int*)take(BB * 2 * 4);
  float*  score = (float*)take((size_t)BB * RR * 4);
  int*    rfirst = (int*)take((size_t)BB * SS * 4);
  int*    rcnt   = (int*)take((size_t)BB * SS * 4);
  float*  text_out = (float*)take((size_t)BB * SS * N2H * 4);
  char*   big = take(2ull * BB * SS * N4H * 4);  // 157 MB region
  float*  xpre_f = (float*)big;
  float*  xpre_b = (float*)(big + (size_t)BB * SS * N4H * 4);
  // overlay (used after LSTM completes; fits inside 'big')
  bf16_t* pwbf   = (bf16_t*)big;
  float*  hidden = (float*)(big + (((size_t)BB * SS * K2H * 2 + 255) & ~(size_t)255));
  float*  x1     = hidden + (size_t)BB * SS * N2H;
  float*  x2     = x1 + (size_t)BB * SS * N2H;
  float*  sumx  = (float*)take((size_t)BB * N2H * 4);
  float*  tsc   = (float*)take((size_t)BB * SS * 4);
  float*  alpha = (float*)take((size_t)BB * SS * 4);
  float*  out2h = (float*)take((size_t)BB * N2H * 4);

  // 1) weight prep (bf16, padded, N-major)
  int nblk = (N4HP * KE + 255) / 256;
  prep_w_kernel<<<nblk, 256, 0, stream>>>(Wf_ih, wfih, N4H, EE, N4HP, KE, 0);
  prep_w_kernel<<<nblk, 256, 0, stream>>>(Wf_hh, wfhh, N4H, HH, N4HP, KE, 0);
  prep_w_kernel<<<nblk, 256, 0, stream>>>(Wb_ih, wbih, N4H, EE, N4HP, KE, 0);
  prep_w_kernel<<<nblk, 256, 0, stream>>>(Wb_hh, wbhh, N4H, HH, N4HP, KE, 0);
  nblk = (N2HP * K2H + 255) / 256;
  prep_w_kernel<<<nblk, 256, 0, stream>>>(gc1_W, gc1w, N2H, N2H, N2HP, K2H, 1);
  prep_w_kernel<<<nblk, 256, 0, stream>>>(gc2_W, gc2w, N2H, N2H, N2HP, K2H, 1);
  relm_kernel<<<(46 * NRELD + 255) / 256, 256, 0, stream>>>(rel_embed, bil_W, relM);
  lens_kernel<<<1, 64, 0, stream>>>(text_idx, aspect_idx, left_idx, lensf, lensi);

  // 2) embedding
  embed_kernel<<<(int)(((size_t)BB * SS * KE + 255) / 256), 256, 0, stream>>>(
      text_idx, embed_tab, textbf);

  // 3) LSTM input projections (big WMMA GEMMs), bias folded in
  dim3 g1(BB * SS / 16, (N4HP + 511) / 512);
  gemm_bf16_wmma<<<g1, 256, 0, stream>>>(textbf, KE, wfih, xpre_f, N4H, bf);
  gemm_bf16_wmma<<<g1, 256, 0, stream>>>(textbf, KE, wbih, xpre_b, N4H, bb);

  // 4) LSTM recurrence (persistent blocks, WMMA on LDS-resident h)
  size_t lstm_smem = (size_t)16 * N4H * 4 + 16 * HH * 4 + 16 * KE * 2;
  lstm_rec_kernel<<<dim3(4, 2), 512, lstm_smem, stream>>>(xpre_f, xpre_b, wfhh, wbhh,
                                                          text_out);

  // 5) relation scores + sparse-adj row ranges
  relscore_kernel<<<(BB * RR * 32 + 255) / 256, 256, 0, stream>>>(
      text_out, relM, head_v, behead_v, rel_v, bil_b, score);
  rowinit_kernel<<<(BB * SS + 255) / 256, 256, 0, stream>>>(rfirst, rcnt);
  rowbuild_kernel<<<(BB * RR + 255) / 256, 256, 0, stream>>>(head_v, rfirst, rcnt);

  // 6) GCN layer 1: pos-weight -> WMMA GEMM -> sparse aggregate + relu
  int pblk = (int)(((size_t)BB * SS * K2H + 255) / 256);
  int ablk = (int)(((size_t)BB * SS * N2H + 255) / 256);
  dim3 g2(BB * SS / 16, (N2HP + 511) / 512);
  poswt_kernel<<<pblk, 256, 0, stream>>>(text_out, lensf, pwbf);
  gemm_bf16_wmma<<<g2, 256, 0, stream>>>(pwbf, K2H, gc1w, hidden, N2H, nullptr);
  gcn_agg_kernel<<<ablk, 256, 0, stream>>>(hidden, score, behead_v, rfirst, rcnt,
                                           gc1_b, x1);
  // 7) GCN layer 2
  poswt_kernel<<<pblk, 256, 0, stream>>>(x1, lensf, pwbf);
  gemm_bf16_wmma<<<g2, 256, 0, stream>>>(pwbf, K2H, gc2w, hidden, N2H, nullptr);
  gcn_agg_kernel<<<ablk, 256, 0, stream>>>(hidden, score, behead_v, rfirst, rcnt,
                                           gc2_b, x2);

  // 8) masked-sum attention + FC
  attn_sum_kernel<<<(BB * N2H + 255) / 256, 256, 0, stream>>>(x2, lensi, sumx);
  attn_score_kernel<<<(BB * SS + 255) / 256, 256, 0, stream>>>(sumx, text_out, tsc);
  softmax_kernel<<<BB, SS, 0, stream>>>(tsc, alpha);
  attn_out_kernel<<<(BB * N2H + 255) / 256, 256, 0, stream>>>(alpha, text_out, out2h);
  fc_kernel<<<1, 256, 0, stream>>>(out2h, fc_W, fc_b, out);
}
